// LSTMLayer_4844723110539
// MI455X (gfx1250) — matrix operand order repaired and checked by reading, not compiled, not run on hardware
//
#include <hip/hip_runtime.h>
#include <hip/hip_bf16.h>

// Problem dims (from reference)
#define BB 64
#define TT 512
#define EE 512
#define HH 1024
#define G4H (4 * HH)       // 4096 gate rows
#define KCAT (EE + HH)     // 1536 fused K
#define BH (BB * HH)       // 65536
#define BTH ((size_t)BB * TT * HH)

typedef __attribute__((ext_vector_type(16))) __bf16 v16bf;
typedef __attribute__((ext_vector_type(8)))  __bf16 v8bf;
typedef __attribute__((ext_vector_type(8)))  float  v8f;
typedef __attribute__((ext_vector_type(4)))  unsigned int v4u;
typedef __attribute__((ext_vector_type(8)))  int v8i;
typedef __attribute__((ext_vector_type(4)))  int v4i;

__device__ __forceinline__ float sigmoidf_(float x) { return 1.0f / (1.0f + __expf(-x)); }

// ---------------- prep kernels ----------------

__global__ void k_zero_out_tail(float* __restrict__ out, const float* __restrict__ h0,
                                const float* __restrict__ c0) {
  size_t i = (size_t)blockIdx.x * blockDim.x + threadIdx.x;
  size_t total = BTH + 2 * (size_t)BH;
  if (i >= total) return;
  if (i < BTH)            out[i] = 0.0f;
  else if (i < BTH + BH)  out[i] = h0[i - BTH];
  else                    out[i] = c0[i - BTH - BH];
}

// Wcat[dir][g][k] (bf16, row-major, K=1536): k<E from U_w, else W_w. bias = U_b + W_b.
__global__ void k_prep_w(const float* __restrict__ Ufw, const float* __restrict__ Wfw,
                         const float* __restrict__ Ubw, const float* __restrict__ Wbw,
                         const float* __restrict__ Ufb, const float* __restrict__ Wfb,
                         const float* __restrict__ Ubb, const float* __restrict__ Wbb,
                         __bf16* __restrict__ Wcat, float* __restrict__ bias) {
  size_t i = (size_t)blockIdx.x * blockDim.x + threadIdx.x;
  const size_t perdir = (size_t)G4H * KCAT;
  if (i >= 2 * perdir) return;
  int dir = (int)(i / perdir);
  size_t r = i % perdir;
  int g = (int)(r / KCAT);
  int k = (int)(r % KCAT);
  float v;
  if (dir == 0) v = (k < EE) ? Ufw[(size_t)g * EE + k] : Wfw[(size_t)g * HH + (k - EE)];
  else          v = (k < EE) ? Ubw[(size_t)g * EE + k] : Wbw[(size_t)g * HH + (k - EE)];
  Wcat[i] = (__bf16)v;
  if (k == 0) bias[dir * G4H + g] = (dir == 0) ? (Ufb[g] + Wfb[g]) : (Ubb[g] + Wbb[g]);
}

// xt[t][b][e] (bf16) from x[b][t][e] (f32)
__global__ void k_prep_x(const float* __restrict__ x, __bf16* __restrict__ xt) {
  size_t i = (size_t)blockIdx.x * blockDim.x + threadIdx.x;
  if (i >= (size_t)BB * TT * EE) return;
  int e = (int)(i % EE);
  size_t bt = i / EE;
  int t = (int)(bt % TT);
  int b = (int)(bt / TT);
  xt[((size_t)t * BB + b) * EE + e] = (__bf16)x[i];
}

// hbuf: [buf(2)][dir(2)][B*H] bf16, init buf 0. cbuf: [dir][B*H] f32.
__global__ void k_init_state(const float* __restrict__ h0, const float* __restrict__ c0,
                             __bf16* __restrict__ hbuf, float* __restrict__ cbuf) {
  size_t i = (size_t)blockIdx.x * blockDim.x + threadIdx.x;
  if (i >= 2 * (size_t)BH) return;
  size_t j = i % BH;
  hbuf[i] = (__bf16)h0[j];
  cbuf[i] = c0[j];
}

// ---------------- TDM descriptor helpers ----------------

__device__ __forceinline__ void setbits256(unsigned long long* q, int lo, int width,
                                           unsigned long long val) {
  int w = lo >> 6, b = lo & 63;
  q[w] |= val << b;
  if (b + width > 64) q[w + 1] |= val >> (64 - b);
}

// 2D tile (rows x cols, 2-byte elements, row stride in elements) -> packed LDS region.
__device__ __forceinline__ void tdm_load_2d_bf16(unsigned lds_off, const void* gaddr,
                                                 unsigned cols, unsigned rows,
                                                 unsigned long long row_stride) {
  unsigned long long ga = (unsigned long long)(uintptr_t)gaddr;
  v4u g0;
  g0[0] = 1u;                                        // count=1, user descriptor
  g0[1] = lds_off;                                   // lds_addr (bytes)
  g0[2] = (unsigned)(ga & 0xffffffffull);            // global_addr[31:0]
  g0[3] = (unsigned)((ga >> 32) & 0x1ffffffull) | 0x80000000u; // addr[56:32] | type=2

  unsigned long long q[4] = {0ull, 0ull, 0ull, 0ull};
  setbits256(q, 16, 2, 1ull);                        // data_size = 1 -> 2 bytes
  setbits256(q, 48, 32, (unsigned long long)cols);   // tensor_dim0
  setbits256(q, 80, 32, (unsigned long long)rows);   // tensor_dim1
  setbits256(q, 112, 16, (unsigned long long)cols);  // tile_dim0
  setbits256(q, 128, 16, (unsigned long long)rows);  // tile_dim1
  setbits256(q, 160, 48, row_stride);                // tensor_dim0_stride
  v8i g1;
#pragma unroll
  for (int i = 0; i < 8; ++i) g1[i] = (int)((q[i >> 1] >> ((i & 1) * 32)) & 0xffffffffull);

  v4i gz4 = {0, 0, 0, 0};
  v8i gz8 = {0, 0, 0, 0, 0, 0, 0, 0};
  // 6-arg form on this toolchain: (g0, g1, g2, g3, extra, cpol)
  __builtin_amdgcn_tensor_load_to_lds(g0, g1, gz4, gz4, gz8, 0);
}

// ---------------- fused LSTM step ----------------
// 128 blocks x 8 waves. Block = (dir, mtile, ngroup of 4 ntiles); wave = (tile 0..3, ks 0..1).
// TDM stages the shared A rows [x_t | h_{t-1}] (16 x 1536 bf16) into LDS; each wave does half
// of K (24 wmma steps) into 4 gate accumulators; ks=1 partials reduced through LDS into ks=0.
__global__ __launch_bounds__(256, 1) void k_lstm_step(
    const __bf16* __restrict__ xt, const __bf16* __restrict__ Wcat,
    const float* __restrict__ bias, __bf16* __restrict__ hbuf,
    float* __restrict__ cbuf, float* __restrict__ out, int s) {
  __shared__ __align__(128) char smem[49152];
  __bf16* sAx = (__bf16*)smem;             // 16 x 512  bf16 = 16 KB (reused as sRed)
  __bf16* sAh = (__bf16*)(smem + 16384);   // 16 x 1024 bf16 = 32 KB
  float*  sRed = (float*)smem;             // aliases sAx after A is consumed

  const int bx    = blockIdx.x;
  const int dir   = bx >> 6;          // 0..1
  const int mtile = (bx >> 4) & 3;    // 0..3 (batch tiles)
  const int ngrp  = bx & 15;          // 0..15 (groups of 4 hidden tiles)
  const int wave  = threadIdx.x >> 5; // 0..7
  const int lane  = threadIdx.x & 31;
  const int tl    = wave >> 1;        // local tile 0..3
  const int ks    = wave & 1;         // K-split half
  const int ntile = ngrp * 4 + tl;

  const int t = dir ? (TT - 1 - s) : s;
  const __bf16* hprev = hbuf + ((size_t)((s & 1) * 2 + dir)) * BH;
  __bf16*       hnext = hbuf + ((size_t)(((s + 1) & 1) * 2 + dir)) * BH;

  // One wave issues the two 2D TDM loads for the block-shared A tile.
  if (wave == 0) {
    const __bf16* gx = xt + ((size_t)t * BB + mtile * 16) * EE;
    const __bf16* gh = hprev + (size_t)(mtile * 16) * HH;
    tdm_load_2d_bf16((unsigned)(uintptr_t)sAx, gx, EE, 16, (unsigned long long)EE);
    tdm_load_2d_bf16((unsigned)(uintptr_t)sAh, gh, HH, 16, (unsigned long long)HH);
    __builtin_amdgcn_s_wait_tensorcnt(0);
  }
  __syncthreads();

  const int col = lane & 15;                 // N within tile / A row within tile
  const int akb = (lane < 16) ? 0 : 8;       // A frag lo-chunk K base (ISA layout)
  const int bkb = (lane < 16) ? 0 : 8;       // B frag lo-chunk K base (ISA layout)

  const __bf16* wdir = Wcat + (size_t)dir * G4H * KCAT;
  const int nglob = ntile * 16 + col;
  const __bf16* wr0 = wdir + ((size_t)(0 * HH + nglob)) * KCAT;  // f
  const __bf16* wr1 = wdir + ((size_t)(1 * HH + nglob)) * KCAT;  // i
  const __bf16* wr2 = wdir + ((size_t)(2 * HH + nglob)) * KCAT;  // g
  const __bf16* wr3 = wdir + ((size_t)(3 * HH + nglob)) * KCAT;  // o
  __builtin_prefetch(wr0 + ks * 768, 0, 1);  // global_prefetch_b8: warm weight rows

  const __bf16* aRowX = sAx + (size_t)col * EE;  // LDS
  const __bf16* aRowH = sAh + (size_t)col * HH;  // LDS

  v8f acc0 = {}, acc1 = {}, acc2 = {}, acc3 = {};

#define GATE_WMMA(W, ACC)                                                              \
  do {                                                                                 \
    v8bf _bl = *(const v8bf*)((W) + k0 + bkb);                                         \
    v8bf _bh = *(const v8bf*)((W) + k0 + bkb + 16);                                    \
    v16bf _b = __builtin_shufflevector(_bl, _bh, 0, 1, 2, 3, 4, 5, 6, 7, 8, 9, 10, 11, \
                                       12, 13, 14, 15);                                \
    ACC = __builtin_amdgcn_wmma_f32_16x16x32_bf16(false, _a, false, _b, (short)0, ACC, \
                                                  false, false);                       \
  } while (0)

  auto kstep = [&](const __bf16* asrc, int k0) {
    v8bf _al = *(const v8bf*)(asrc + akb);        // ds_load_b128
    v8bf _ah = *(const v8bf*)(asrc + akb + 16);   // ds_load_b128
    v16bf _a = __builtin_shufflevector(_al, _ah, 0, 1, 2, 3, 4, 5, 6, 7, 8, 9, 10, 11,
                                       12, 13, 14, 15);
    GATE_WMMA(wr0, acc0);
    GATE_WMMA(wr1, acc1);
    GATE_WMMA(wr2, acc2);
    GATE_WMMA(wr3, acc3);
  };

  if (ks == 0) {  // K = 0..767: x part (16 iters) + first h part (8 iters)
#pragma unroll 4
    for (int kk = 0; kk < 16; ++kk) { int k0 = kk * 32; kstep(aRowX + k0, k0); }
#pragma unroll 4
    for (int kk = 16; kk < 24; ++kk) { int k0 = kk * 32; kstep(aRowH + (k0 - EE), k0); }
  } else {        // K = 768..1535: h part (24 iters)
#pragma unroll 4
    for (int kk = 24; kk < 48; ++kk) { int k0 = kk * 32; kstep(aRowH + (k0 - EE), k0); }
  }
#undef GATE_WMMA

  // Cross-wave K reduction through LDS (sRed aliases sAx: A fully consumed by now).
  __syncthreads();
  float* slot = sRed + ((size_t)tl * 1024 + (size_t)lane * 32);
  if (ks == 1) {
    *(v8f*)(slot + 0)  = acc0;
    *(v8f*)(slot + 8)  = acc1;
    *(v8f*)(slot + 16) = acc2;
    *(v8f*)(slot + 24) = acc3;
  }
  __syncthreads();
  if (ks == 0) {
    acc0 += *(const v8f*)(slot + 0);
    acc1 += *(const v8f*)(slot + 8);
    acc2 += *(const v8f*)(slot + 16);
    acc3 += *(const v8f*)(slot + 24);

    // Epilogue: gate nonlinearity + cell update.
    const float bF = bias[dir * G4H + 0 * HH + nglob];
    const float bI = bias[dir * G4H + 1 * HH + nglob];
    const float bG = bias[dir * G4H + 2 * HH + nglob];
    const float bO = bias[dir * G4H + 3 * HH + nglob];
    float* cdir = cbuf + (size_t)dir * BH;
    const int mofs = (lane < 16) ? 0 : 8;  // C/D layout: VGPR r -> M = mofs + r

#pragma unroll
    for (int r = 0; r < 8; ++r) {
      const int b = mtile * 16 + mofs + r;
      const size_t sidx = (size_t)b * HH + nglob;
      const float fv = sigmoidf_(acc0[r] + bF);
      const float iv = sigmoidf_(acc1[r] + bI);
      const float gv = tanhf(acc2[r] + bG);
      const float ov = sigmoidf_(acc3[r] + bO);
      const float cn = cdir[sidx] * fv + iv * gv;
      cdir[sidx] = cn;
      const float hv = ov * tanhf(cn);
      hnext[sidx] = (__bf16)hv;
      out[((size_t)b * TT + t) * HH + nglob] += hv;  // fwd/bwd hit row in different launches
    }
  }
}

// ---------------- host launcher ----------------
extern "C" void kernel_launch(void* const* d_in, const int* in_sizes, int n_in,
                              void* d_out, int out_size, void* d_ws, size_t ws_size,
                              hipStream_t stream) {
  (void)in_sizes; (void)n_in; (void)out_size; (void)ws_size;
  const float* x   = (const float*)d_in[0];
  const float* h0  = (const float*)d_in[1];
  const float* c0  = (const float*)d_in[2];
  const float* Ufw = (const float*)d_in[3];
  const float* Ufb = (const float*)d_in[4];
  const float* Wfw = (const float*)d_in[5];
  const float* Wfb = (const float*)d_in[6];
  const float* Ubw = (const float*)d_in[7];
  const float* Ubb = (const float*)d_in[8];
  const float* Wbw = (const float*)d_in[9];
  const float* Wbb = (const float*)d_in[10];
  float* out = (float*)d_out;

  char* ws = (char*)d_ws;
  __bf16* xt   = (__bf16*)ws;  ws += (size_t)TT * BB * EE * sizeof(__bf16);      // 33.5 MB
  __bf16* Wcat = (__bf16*)ws;  ws += (size_t)2 * G4H * KCAT * sizeof(__bf16);    // 25.2 MB
  float*  bias = (float*)ws;   ws += (size_t)2 * G4H * sizeof(float);
  __bf16* hbuf = (__bf16*)ws;  ws += (size_t)2 * 2 * BH * sizeof(__bf16);        // double-buffered
  float*  cbuf = (float*)ws;   ws += (size_t)2 * BH * sizeof(float);

  {
    size_t total = BTH + 2 * (size_t)BH;
    k_zero_out_tail<<<(unsigned)((total + 255) / 256), 256, 0, stream>>>(out, h0, c0);
  }
  {
    size_t total = (size_t)2 * G4H * KCAT;
    k_prep_w<<<(unsigned)((total + 255) / 256), 256, 0, stream>>>(
        Ufw, Wfw, Ubw, Wbw, Ufb, Wfb, Ubb, Wbb, Wcat, bias);
  }
  {
    size_t total = (size_t)BB * TT * EE;
    k_prep_x<<<(unsigned)((total + 255) / 256), 256, 0, stream>>>(x, xt);
  }
  {
    size_t total = 2 * (size_t)BH;
    k_init_state<<<(unsigned)((total + 255) / 256), 256, 0, stream>>>(h0, c0, hbuf, cbuf);
  }
  // Sequential scan: 512 launches, fwd and bwd fused per launch.
  for (int s = 0; s < TT; ++s) {
    k_lstm_step<<<128, 256, 0, stream>>>(xt, Wcat, bias, hbuf, cbuf, out, s);
  }
}
